// DynamicHeterogeneousGraphEncoder_55697135895021
// MI455X (gfx1250) — matrix operand order repaired and checked by reading, not compiled, not run on hardware
//
#include <hip/hip_runtime.h>

// Problem constants from the reference (N, E derived from in_sizes at launch).
#define H_DIM   128
#define D_DIM   128
#define ED_DIM  64
#define T_TYPES 2
#define L_LEVELS 4

typedef unsigned int u32;
typedef __attribute__((ext_vector_type(8)))  float   v8f;
typedef __attribute__((ext_vector_type(16))) __bf16  v16bf;
typedef __attribute__((ext_vector_type(4)))  float   v4f;
typedef __attribute__((ext_vector_type(2)))  u32     v2u;
typedef __attribute__((ext_vector_type(4)))  u32     v4u;

union AccU  { v8f v; float e[8]; };
union FragU { v16bf v; __bf16 e[16]; u32 u[8]; v4u q[2]; };

// ---------- small helpers ----------
__device__ inline unsigned short f2bf(float f) {
  u32 u = __float_as_uint(f);
  u += 0x7FFFu + ((u >> 16) & 1u);          // round-to-nearest-even
  return (unsigned short)(u >> 16);
}
__device__ inline u32 pack_bf2(float lo, float hi) {   // {K even, K odd} in one dword
  return (u32)f2bf(lo) | ((u32)f2bf(hi) << 16);
}
// order-preserving float -> uint map for atomicMax-based segment max
__device__ inline u32 enc_f(float f) {
  u32 u = __float_as_uint(f);
  return (u & 0x80000000u) ? ~u : (u | 0x80000000u);
}
__device__ inline float dec_f(u32 e) {
  u32 u = (e & 0x80000000u) ? (e & 0x7FFFFFFFu) : ~e;
  return __uint_as_float(u);                 // dec(0) == NaN == "empty segment"
}

// ---------- zero fill ----------
__global__ void zero_f32_kernel(float* p, long n) {
  long i = (long)blockIdx.x * 256 + threadIdx.x;
  if (i < n) p[i] = 0.f;
}

// =====================================================================
// Generic WMMA GEMM:  C[M x 128] = act( A[M x K](f32) @ W[K x 128](f32) + bias )
// K must be a multiple of 32 (true at every call site: 128).
// LDS layout is packed bf16 K-pairs (u32), B stored transposed, so every
// WMMA fragment load is two contiguous 16B chunks -> ds_load_b128.
// Block = 256 thr = 8 waves; tile = 32 rows x 128 cols; 2 16x16 tiles/wave.
// =====================================================================
__global__ __launch_bounds__(256) void gemm_bias_act_kernel(
    const float* __restrict__ A, int lda,
    const float* __restrict__ W,            // K x 128 row-major
    const float* __restrict__ bias,         // 128 or nullptr
    float* __restrict__ C,                  // M x 128
    int M, int K, int act,                  // act: 0 none, 1 tanh
    const int* __restrict__ levels, int lsel)
{
  __shared__ __align__(16) u32 As[32][20];   // [row][kpair]  (16 used + pad)
  __shared__ __align__(16) u32 Bt[128][20];  // [col][kpair]  (transposed W chunk)
  const int tid  = threadIdx.x;
  const int lane = tid & 31;
  const int wave = tid >> 5;
  const int rowHalf = wave & 1;
  const int colPair = wave >> 1;            // 0..3 -> cols 32*colPair
  const int row0 = blockIdx.x * 32;

  // staging roles
  const int ar  = tid >> 3;                 // A row 0..31
  const int ac4 = (tid & 7) * 4;            // A col group (4 floats)
  const int bj  = tid & 15;                 // B k-pair index 0..15
  const int bc8 = (tid >> 4) * 8;           // B col group (8 cols)

  AccU acc0, acc1;
  for (int i = 0; i < 8; ++i) { acc0.e[i] = 0.f; acc1.e[i] = 0.f; }

  for (int k0 = 0; k0 < K; k0 += 32) {
    { // stage A: one b128 global load + one b64 LDS store per thread
      v4f a = {0.f, 0.f, 0.f, 0.f};
      int gr = row0 + ar;
      if (gr < M) a = *(const v4f*)&A[(size_t)gr * lda + k0 + ac4];
      v2u p; p.x = pack_bf2(a.x, a.y); p.y = pack_bf2(a.z, a.w);
      *(v2u*)&As[ar][ac4 >> 1] = p;
    }
    { // stage B transposed: Bt[c][j] = {W[k0+2j][c], W[k0+2j+1][c]}
      const float* w0 = &W[(size_t)(k0 + 2 * bj) * 128 + bc8];
      v4f a0 = *(const v4f*)w0,         a1 = *(const v4f*)(w0 + 4);
      v4f b0 = *(const v4f*)(w0 + 128), b1 = *(const v4f*)(w0 + 132);
      Bt[bc8 + 0][bj] = pack_bf2(a0.x, b0.x);
      Bt[bc8 + 1][bj] = pack_bf2(a0.y, b0.y);
      Bt[bc8 + 2][bj] = pack_bf2(a0.z, b0.z);
      Bt[bc8 + 3][bj] = pack_bf2(a0.w, b0.w);
      Bt[bc8 + 4][bj] = pack_bf2(a1.x, b1.x);
      Bt[bc8 + 5][bj] = pack_bf2(a1.y, b1.y);
      Bt[bc8 + 6][bj] = pack_bf2(a1.z, b1.z);
      Bt[bc8 + 7][bj] = pack_bf2(a1.w, b1.w);
    }
    __syncthreads();

    FragU af, bf0, bf1;
    { // A fragment: lanes 0-15 M=lane K0..15 pairs; lanes 16-31 same M, +8 K offset
      int m = lane & 15, h = lane >> 4;
      const u32* rowp = As[rowHalf * 16 + m];
      af.q[0] = *(const v4u*)&rowp[h * 4];
      af.q[1] = *(const v4u*)&rowp[8 + h * 4];
    }
    { // B fragments: lane n holds col, kh adds +8 K offset (same K order as A)
      int n = lane & 15, kh = lane >> 4;
      const u32* p0 = Bt[colPair * 32 + n];
      const u32* p1 = Bt[colPair * 32 + n + 16];
      bf0.q[0] = *(const v4u*)&p0[kh * 4];
      bf0.q[1] = *(const v4u*)&p0[8 + kh * 4];
      bf1.q[0] = *(const v4u*)&p1[kh * 4];
      bf1.q[1] = *(const v4u*)&p1[8 + kh * 4];
    }
    acc0.v = __builtin_amdgcn_wmma_f32_16x16x32_bf16(false, af.v, false, bf0.v,
                                                     (short)0, acc0.v, false, false);
    acc1.v = __builtin_amdgcn_wmma_f32_16x16x32_bf16(false, af.v, false, bf1.v,
                                                     (short)0, acc1.v, false, false);
    __syncthreads();
  }

  int half = lane >> 4, n = lane & 15;
  for (int v = 0; v < 8; ++v) {
    int gr = row0 + rowHalf * 16 + v + 8 * half;
    if (gr >= M) continue;
    if (levels && levels[gr] != lsel) continue;
    int c0 = colPair * 32 + n, c1 = c0 + 16;
    float x0 = acc0.e[v], x1 = acc1.e[v];
    if (bias) { x0 += bias[c0]; x1 += bias[c1]; }
    if (act == 1) { x0 = tanhf(x0); x1 = tanhf(x1); }
    C[(size_t)gr * 128 + c0] = x0;
    C[(size_t)gr * 128 + c1] = x1;
  }
}

// =====================================================================
// Fused per-edge attention score:
//   pre[e,:] = h[s[e]]@W1a + h[d[e]]@W1b + ef[e]@Wc        (K = 128+128+64)
//   a[e]     = sum_c tanh(pre + bc[c]) * W2[c] + b2
// plus atomicMax segment-max into mbits[s[e]].
// Same packed-LDS WMMA scheme as the GEMM; A rows are gathered per edge.
// =====================================================================
__global__ __launch_bounds__(256) void edge_att_kernel(
    const float* __restrict__ h,            // N x 128
    const float* __restrict__ ef,           // E x 64
    const int*   __restrict__ src,
    const int*   __restrict__ dst,
    const float* __restrict__ W1,           // 384 x 128 (W1a rows 0..127, W1b 128..255)
    const float* __restrict__ Wc,           // 64 x 128  (= We @ W1c, pre-folded)
    const float* __restrict__ bc,           // 128       (= be @ W1c + b1)
    const float* __restrict__ W2,           // 128
    const float* __restrict__ b2p,          // scalar
    float* __restrict__ ea,                 // E
    u32*   __restrict__ mbits,              // N
    int E)
{
  __shared__ __align__(16) u32 As[32][20];
  __shared__ __align__(16) u32 Bt[128][20];
  __shared__ int   sIdx[32], dIdx[32];
  __shared__ float sAcc[32];

  const int tid  = threadIdx.x;
  const int lane = tid & 31;
  const int wave = tid >> 5;
  const int rowHalf = wave & 1;
  const int colPair = wave >> 1;
  const int e0 = blockIdx.x * 32;

  const int ar  = tid >> 3;
  const int ac4 = (tid & 7) * 4;
  const int bj  = tid & 15;
  const int bc8 = (tid >> 4) * 8;

  if (tid < 32) {
    int e = e0 + tid;
    sIdx[tid] = (e < E) ? src[e] : -1;
    dIdx[tid] = (e < E) ? dst[e] : -1;
    sAcc[tid] = 0.f;
  }
  __syncthreads();

  AccU acc0, acc1;
  for (int i = 0; i < 8; ++i) { acc0.e[i] = 0.f; acc1.e[i] = 0.f; }

  for (int phase = 0; phase < 3; ++phase) {
    const float* Bp = (phase == 0) ? W1 : (phase == 1 ? W1 + 128 * 128 : Wc);
    const int Kp = (phase == 2) ? 64 : 128;
    for (int k0 = 0; k0 < Kp; k0 += 32) {
      { // gather-stage A: one b128 global load per thread
        v4f a = {0.f, 0.f, 0.f, 0.f};
        if (phase < 2) {
          int node = (phase == 0) ? sIdx[ar] : dIdx[ar];
          if (node >= 0) a = *(const v4f*)&h[(size_t)node * 128 + k0 + ac4];
        } else {
          int e = e0 + ar;
          if (e < E) a = *(const v4f*)&ef[(size_t)e * 64 + k0 + ac4];
        }
        v2u p; p.x = pack_bf2(a.x, a.y); p.y = pack_bf2(a.z, a.w);
        *(v2u*)&As[ar][ac4 >> 1] = p;
      }
      { // stage B transposed
        const float* w0 = &Bp[(size_t)(k0 + 2 * bj) * 128 + bc8];
        v4f a0 = *(const v4f*)w0,         a1 = *(const v4f*)(w0 + 4);
        v4f b0 = *(const v4f*)(w0 + 128), b1 = *(const v4f*)(w0 + 132);
        Bt[bc8 + 0][bj] = pack_bf2(a0.x, b0.x);
        Bt[bc8 + 1][bj] = pack_bf2(a0.y, b0.y);
        Bt[bc8 + 2][bj] = pack_bf2(a0.z, b0.z);
        Bt[bc8 + 3][bj] = pack_bf2(a0.w, b0.w);
        Bt[bc8 + 4][bj] = pack_bf2(a1.x, b1.x);
        Bt[bc8 + 5][bj] = pack_bf2(a1.y, b1.y);
        Bt[bc8 + 6][bj] = pack_bf2(a1.z, b1.z);
        Bt[bc8 + 7][bj] = pack_bf2(a1.w, b1.w);
      }
      __syncthreads();

      FragU af, bf0, bf1;
      {
        int m = lane & 15, hh = lane >> 4;
        const u32* rowp = As[rowHalf * 16 + m];
        af.q[0] = *(const v4u*)&rowp[hh * 4];
        af.q[1] = *(const v4u*)&rowp[8 + hh * 4];
      }
      {
        int n = lane & 15, kh = lane >> 4;
        const u32* p0 = Bt[colPair * 32 + n];
        const u32* p1 = Bt[colPair * 32 + n + 16];
        bf0.q[0] = *(const v4u*)&p0[kh * 4];
        bf0.q[1] = *(const v4u*)&p0[8 + kh * 4];
        bf1.q[0] = *(const v4u*)&p1[kh * 4];
        bf1.q[1] = *(const v4u*)&p1[8 + kh * 4];
      }
      acc0.v = __builtin_amdgcn_wmma_f32_16x16x32_bf16(false, af.v, false, bf0.v,
                                                       (short)0, acc0.v, false, false);
      acc1.v = __builtin_amdgcn_wmma_f32_16x16x32_bf16(false, af.v, false, bf1.v,
                                                       (short)0, acc1.v, false, false);
      __syncthreads();
    }
  }

  // tanh(pre + bc) * W2, reduce over 128 columns into per-edge score (LDS atomics)
  {
    int half = lane >> 4, n = lane & 15;
    for (int v = 0; v < 8; ++v) {
      int lr = rowHalf * 16 + v + 8 * half;
      int c0 = colPair * 32 + n, c1 = c0 + 16;
      atomicAdd(&sAcc[lr], tanhf(acc0.e[v] + bc[c0]) * W2[c0]);
      atomicAdd(&sAcc[lr], tanhf(acc1.e[v] + bc[c1]) * W2[c1]);
    }
  }
  __syncthreads();
  if (tid < 32) {
    int e = e0 + tid;
    if (e < E) {
      float a = sAcc[tid] + b2p[0];
      ea[e] = a;
      atomicMax(&mbits[sIdx[tid]], enc_f(a));
    }
  }
}

// ---------- segmented softmax pass 2: exp + sum ----------
__global__ void seg_exp_kernel(float* __restrict__ ea, const int* __restrict__ src,
                               const u32* __restrict__ mbits,
                               float* __restrict__ z, int E) {
  int e = blockIdx.x * 256 + threadIdx.x;
  if (e >= E) return;
  int s = src[e];
  float m = dec_f(mbits[s]);
  if (!__builtin_isfinite(m)) m = 0.f;      // matches where(isfinite(m), m, 0)
  float v = __expf(ea[e] - m);
  ea[e] = v;
  atomicAdd(&z[s], v);
}

// ---------- agg[s] += alpha * h[d] ----------
__global__ void seg_agg_kernel(const float* __restrict__ ea, const int* __restrict__ src,
                               const int* __restrict__ dst, const float* __restrict__ z,
                               const float* __restrict__ h, float* __restrict__ agg, int E) {
  long tid = (long)blockIdx.x * 256 + threadIdx.x;
  if (tid >= (long)E * 128) return;
  int e = (int)(tid >> 7), c = (int)(tid & 127);
  int s = src[e], d = dst[e];
  float alpha = ea[e] / (z[s] + 1e-16f);
  atomicAdd(&agg[(size_t)s * 128 + c], alpha * h[(size_t)d * 128 + c]);
}

// ---------- row LayerNorm + ReLU (wave32 shuffle reduction, 1 wave/row) ----------
__global__ __launch_bounds__(256) void ln_relu_kernel(
    const float* __restrict__ x, const float* __restrict__ g,
    const float* __restrict__ b, float* __restrict__ y, int M) {
  int wave = threadIdx.x >> 5, lane = threadIdx.x & 31;
  int row = blockIdx.x * 8 + wave;
  if (row >= M) return;
  float v[4], s = 0.f, ss = 0.f;
  for (int i = 0; i < 4; ++i) {
    v[i] = x[(size_t)row * 128 + lane + 32 * i];
    s += v[i]; ss += v[i] * v[i];
  }
  for (int m = 16; m >= 1; m >>= 1) { s += __shfl_xor(s, m); ss += __shfl_xor(ss, m); }
  float mean = s * (1.f / 128.f);
  float var  = ss * (1.f / 128.f) - mean * mean;
  float inv  = rsqrtf(var + 1e-5f);
  for (int i = 0; i < 4; ++i) {
    int c = lane + 32 * i;
    float o = (v[i] - mean) * inv * g[c] + b[c];
    y[(size_t)row * 128 + c] = fmaxf(o, 0.f);
  }
}

// ---------- out[row*stride+off] = dot(X[row], w) + b ----------
__global__ __launch_bounds__(256) void rowdot_kernel(
    const float* __restrict__ X, const float* __restrict__ w,
    const float* __restrict__ bp, float* __restrict__ out,
    int M, int stride, int off) {
  int wave = threadIdx.x >> 5, lane = threadIdx.x & 31;
  int row = blockIdx.x * 8 + wave;
  if (row >= M) return;
  float s = 0.f;
  for (int i = 0; i < 4; ++i) { int c = lane + 32 * i; s += X[(size_t)row * 128 + c] * w[c]; }
  for (int m = 16; m >= 1; m >>= 1) s += __shfl_xor(s, m);
  if (lane == 0) out[(size_t)row * stride + off] = s + bp[0];
}

// ---------- softmax(scores[n,0..1]) blend of the two type embeddings ----------
__global__ void fuse_kernel(const float* __restrict__ emb, const float* __restrict__ scores,
                            float* __restrict__ out, int M) {
  long tid = (long)blockIdx.x * 256 + threadIdx.x;
  if (tid >= (long)M * 128) return;
  int n = (int)(tid >> 7), c = (int)(tid & 127);
  float s0 = scores[(size_t)n * 2], s1 = scores[(size_t)n * 2 + 1];
  float mx = fmaxf(s0, s1);
  float e0 = __expf(s0 - mx), e1 = __expf(s1 - mx);
  float inv = 1.f / (e0 + e1);
  out[tid] = (e0 * inv) * emb[(size_t)n * 128 + c]
           + (e1 * inv) * emb[(size_t)(M + n) * 128 + c];
}

// ---------- bc = be @ W1c + b1 ----------
__global__ void bc_kernel(const float* __restrict__ be, const float* __restrict__ W1c,
                          const float* __restrict__ b1, float* __restrict__ bc) {
  int j = threadIdx.x;                      // 128 threads
  float s = 0.f;
  for (int k = 0; k < 128; ++k) s += be[k] * W1c[(size_t)k * 128 + j];
  bc[j] = s + b1[j];
}

// ---------- per-level feature sum + count ----------
__global__ void lmean_acc_kernel(const float* __restrict__ hf, const int* __restrict__ levels,
                                 float* __restrict__ lraw, float* __restrict__ counts, int M) {
  long tid = (long)blockIdx.x * 256 + threadIdx.x;
  if (tid >= (long)M * 128) return;
  int n = (int)(tid >> 7), c = (int)(tid & 127);
  int l = levels[n];
  atomicAdd(&lraw[l * 128 + c], hf[tid]);
  if (c == 0) atomicAdd(&counts[l], 1.f);
}
__global__ void lmean_fin_kernel(const float* __restrict__ lraw, const float* __restrict__ counts,
                                 float* __restrict__ lmean) {
  int tid = threadIdx.x;                    // 512
  int l = tid >> 7;
  lmean[tid] = lraw[tid] / fmaxf(counts[l], 1.f);
}

// ---------- vl[l] = lmean[l] @ W1b ----------
__global__ void vl_kernel(const float* __restrict__ lmean, const float* __restrict__ W1b,
                          float* __restrict__ vl) {
  int tid = threadIdx.x;                    // 512: l = tid>>7, j = tid&127
  int l = tid >> 7, j = tid & 127;
  float s = 0.f;
  for (int k = 0; k < 128; ++k) s += lmean[l * 128 + k] * W1b[(size_t)k * 128 + j];
  vl[tid] = s;
}

// ---------- hierarchical attention over L=4 levels (1 wave / node) ----------
__global__ __launch_bounds__(256) void hier_att_kernel(
    const float* __restrict__ u, const float* __restrict__ vl,
    const float* __restrict__ b1, const float* __restrict__ W2,
    const float* __restrict__ b2p, const float* __restrict__ lmean,
    float* __restrict__ out, int M) {
  int wave = threadIdx.x >> 5, lane = threadIdx.x & 31;
  int row = blockIdx.x * 8 + wave;
  if (row >= M) return;
  float x[4];
  for (int i = 0; i < 4; ++i) x[i] = u[(size_t)row * 128 + lane + 32 * i];
  float sc[4];
  for (int l = 0; l < 4; ++l) {
    float p = 0.f;
    for (int i = 0; i < 4; ++i) {
      int c = lane + 32 * i;
      p += tanhf(x[i] + vl[l * 128 + c] + b1[c]) * W2[c];
    }
    for (int m = 16; m >= 1; m >>= 1) p += __shfl_xor(p, m);
    sc[l] = p + b2p[0];
  }
  float mx = fmaxf(fmaxf(sc[0], sc[1]), fmaxf(sc[2], sc[3]));
  float ew[4], ssum = 0.f;
  for (int l = 0; l < 4; ++l) { ew[l] = __expf(sc[l] - mx); ssum += ew[l]; }
  float inv = 1.f / ssum;
  for (int i = 0; i < 4; ++i) {
    int c = lane + 32 * i;
    float o = 0.f;
    for (int l = 0; l < 4; ++l) o += ew[l] * inv * lmean[l * 128 + c];
    out[(size_t)row * 128 + c] = o;
  }
}

// =====================================================================
extern "C" void kernel_launch(void* const* d_in, const int* in_sizes, int n_in,
                              void* d_out, int out_size, void* d_ws, size_t ws_size,
                              hipStream_t stream) {
  (void)n_in; (void)out_size; (void)ws_size;
  const float* node_features = (const float*)d_in[0];
  const float* edge_features = (const float*)d_in[1];
  const int*   node_levels   = (const int*)d_in[2];
  const int*   edge_index    = (const int*)d_in[3];
  const float* nl_node_W = (const float*)d_in[4];
  const float* nl_node_b = (const float*)d_in[5];
  const float* nl_edge_W = (const float*)d_in[6];
  const float* nl_edge_b = (const float*)d_in[7];
  const float* nl_att_W1 = (const float*)d_in[8];
  const float* nl_att_b1 = (const float*)d_in[9];
  const float* nl_att_W2 = (const float*)d_in[10];
  const float* nl_att_b2 = (const float*)d_in[11];
  const float* nl_out_W  = (const float*)d_in[12];
  const float* nl_out_b  = (const float*)d_in[13];
  const float* nl_ln_g   = (const float*)d_in[14];
  const float* nl_ln_b   = (const float*)d_in[15];
  const float* el_att_W1 = (const float*)d_in[16];
  const float* el_att_b1 = (const float*)d_in[17];
  const float* el_att_W2 = (const float*)d_in[18];
  const float* el_att_b2 = (const float*)d_in[19];
  const float* el_out_W  = (const float*)d_in[20];
  const float* el_out_b  = (const float*)d_in[21];
  const float* el_ln_g   = (const float*)d_in[22];
  const float* el_ln_b   = (const float*)d_in[23];
  const float* hl_proj_W = (const float*)d_in[24];
  const float* hl_proj_b = (const float*)d_in[25];
  const float* hi_att_W1 = (const float*)d_in[26];
  const float* hi_att_b1 = (const float*)d_in[27];
  const float* hi_att_W2 = (const float*)d_in[28];
  const float* hi_att_b2 = (const float*)d_in[29];
  const float* hl_out_W  = (const float*)d_in[30];
  const float* hl_out_b  = (const float*)d_in[31];
  const float* hl_ln_g   = (const float*)d_in[32];
  const float* hl_ln_b   = (const float*)d_in[33];

  const int N = in_sizes[0] / D_DIM;                   // 100000
  const int E = in_sizes[1] / (T_TYPES * ED_DIM);      // 500000

  // workspace layout (all f32)
  float* ws = (float*)d_ws;
  size_t off = 0;
  auto take = [&](size_t n) { float* p = ws + off; off += n; return p; };
  float* hbuf  = take((size_t)N * 128);
  float* emb   = take((size_t)2 * N * 128);            // emb0 | emb1
  float* aggb  = take((size_t)N * 128);
  float* tmp   = take((size_t)N * 128);
  float* tmp2  = take((size_t)N * 128);
  float* hf    = take((size_t)N * 128);
  float* ea    = take((size_t)E);
  float* z     = take((size_t)N);
  u32*   mbits = (u32*)take((size_t)N);
  float* scores = take((size_t)2 * N);
  float* Wc    = take(64 * 128);
  float* bc    = take(128);
  float* lraw  = take(4 * 128);
  float* counts = take(4);
  float* lmean = take(4 * 128);
  float* vl    = take(4 * 128);

  const dim3 blk(256);
  const unsigned gemmBlocks = (unsigned)((N + 31) / 32);
  const unsigned rowBlocks  = (unsigned)((N + 7) / 8);
  const unsigned nc128Blocks = (unsigned)(((long)N * 128 + 255) / 256);
  auto zerof = [&](float* p, long n) {
    zero_f32_kernel<<<(unsigned)((n + 255) / 256), blk, 0, stream>>>(p, n);
  };

  zerof(lraw, 512);
  zerof(counts, 4);

  for (int t = 0; t < T_TYPES; ++t) {
    const float* W1 = nl_att_W1 + (size_t)t * 384 * 128;
    // h = node_features @ nl_node_W[t] + b
    gemm_bias_act_kernel<<<gemmBlocks, blk, 0, stream>>>(
        node_features, 128, nl_node_W + (size_t)t * 128 * 128,
        nl_node_b + t * 128, hbuf, N, 128, 0, nullptr, 0);
    // Wc = We @ W1c  (folds the edge projection into the attention MLP)
    gemm_bias_act_kernel<<<2, blk, 0, stream>>>(
        nl_edge_W + (size_t)t * 64 * 128, 128, W1 + 256 * 128,
        nullptr, Wc, 64, 128, 0, nullptr, 0);
    bc_kernel<<<1, 128, 0, stream>>>(nl_edge_b + t * 128, W1 + 256 * 128,
                                     nl_att_b1 + t * 128, bc);
    zerof(z, N);
    zerof((float*)mbits, N);
    zerof(aggb, (long)N * 128);

    const int* src = edge_index + (size_t)t * 2 * E;
    const int* dst = src + E;
    edge_att_kernel<<<(unsigned)((E + 31) / 32), blk, 0, stream>>>(
        hbuf, edge_features + (size_t)t * E * 64, src, dst,
        W1, Wc, bc, nl_att_W2 + t * 128, nl_att_b2 + t, ea, mbits, E);
    seg_exp_kernel<<<(unsigned)((E + 255) / 256), blk, 0, stream>>>(ea, src, mbits, z, E);
    seg_agg_kernel<<<(unsigned)(((long)E * 128 + 255) / 256), blk, 0, stream>>>(
        ea, src, dst, z, hbuf, aggb, E);

    // emb[t] = relu(LN(agg @ out_W + b))
    gemm_bias_act_kernel<<<gemmBlocks, blk, 0, stream>>>(
        aggb, 128, nl_out_W + (size_t)t * 128 * 128,
        nl_out_b + t * 128, tmp, N, 128, 0, nullptr, 0);
    ln_relu_kernel<<<rowBlocks, blk, 0, stream>>>(
        tmp, nl_ln_g + t * 128, nl_ln_b + t * 128, emb + (size_t)t * N * 128, N);

    // scores[:,t] = tanh(emb @ W1 + b1) @ W2 + b2
    gemm_bias_act_kernel<<<gemmBlocks, blk, 0, stream>>>(
        emb + (size_t)t * N * 128, 128, el_att_W1 + (size_t)t * 128 * 128,
        el_att_b1 + t * 128, tmp, N, 128, 1 /*tanh*/, nullptr, 0);
    rowdot_kernel<<<rowBlocks, blk, 0, stream>>>(
        tmp, el_att_W2 + t * 128, el_att_b2 + t, scores, N, 2, t);
  }

  // fused = relu(LN(softmax-blend(emb) @ el_out_W + b))
  fuse_kernel<<<nc128Blocks, blk, 0, stream>>>(emb, scores, tmp2, N);
  gemm_bias_act_kernel<<<gemmBlocks, blk, 0, stream>>>(
      tmp2, 128, el_out_W, el_out_b, tmp, N, 128, 0, nullptr, 0);
  ln_relu_kernel<<<rowBlocks, blk, 0, stream>>>(tmp, el_ln_g, el_ln_b, tmp2, N);

  // hf[n] = fused[n] @ hl_proj_W[level[n]] + b  (level-masked stores, 4 passes)
  for (int l = 0; l < L_LEVELS; ++l)
    gemm_bias_act_kernel<<<gemmBlocks, blk, 0, stream>>>(
        tmp2, 128, hl_proj_W + (size_t)l * 128 * 128,
        hl_proj_b + l * 128, hf, N, 128, 0, node_levels, l);

  lmean_acc_kernel<<<nc128Blocks, blk, 0, stream>>>(hf, node_levels, lraw, counts, N);
  lmean_fin_kernel<<<1, 512, 0, stream>>>(lraw, counts, lmean);

  // u = hf @ W1a ;  vl = lmean @ W1b
  gemm_bias_act_kernel<<<gemmBlocks, blk, 0, stream>>>(
      hf, 128, hi_att_W1, nullptr, tmp, N, 128, 0, nullptr, 0);
  vl_kernel<<<1, 512, 0, stream>>>(lmean, hi_att_W1 + 128 * 128, vl);

  hier_att_kernel<<<rowBlocks, blk, 0, stream>>>(
      tmp, vl, hi_att_b1, hi_att_W2, hi_att_b2, lmean, tmp2, N);

  // out = relu(LN(enhanced @ hl_out_W + b))
  gemm_bias_act_kernel<<<gemmBlocks, blk, 0, stream>>>(
      tmp2, 128, hl_out_W, hl_out_b, tmp, N, 128, 0, nullptr, 0);
  ln_relu_kernel<<<rowBlocks, blk, 0, stream>>>(tmp, hl_ln_g, hl_ln_b, (float*)d_out, N);
}
